// SyncGeodesicConv_50019189129838
// MI455X (gfx1250) — matrix operand order repaired and checked by reading, not compiled, not run on hardware
//
#include <hip/hip_runtime.h>

typedef _Float16 v16h __attribute__((ext_vector_type(16)));
typedef _Float16 v8h  __attribute__((ext_vector_type(8)));
typedef _Float16 v4h  __attribute__((ext_vector_type(4)));
typedef _Float16 v2h  __attribute__((ext_vector_type(2)));
typedef float    v8f  __attribute__((ext_vector_type(8)));
typedef float    v4f  __attribute__((ext_vector_type(4)));

__device__ __forceinline__ v2h cvt_pk_h2(float a, float b) {
    return __builtin_bit_cast(v2h, __builtin_amdgcn_cvt_pkrtz(a, b));
}

namespace geo {
constexpr int NB     = 2;
constexpr int NV     = 30000;
constexpr int NDIRS  = 8;
constexpr int NRINGS = 3;
constexpr int NC     = 16;
constexpr int NF     = 16;
constexpr int KCONV  = NRINGS * NDIRS * NC;   // 384
constexpr int KTOT   = KCONV + 32;            // 416 = conv + center(16) + zero pad(16)
constexpr int WAVES  = 8;                     // 256 threads, wave32
constexpr int NPAIRS = (NB * NV) / 2;         // 30000
constexpr int BLOCKS = 750;                   // 6000 wave-slots -> exactly 5 pairs/wave
}
using namespace geo;

__global__ __launch_bounds__(256) void geo_conv_wmma(
    const float* __restrict__ y,      // (B, NV, NDIRS, C)
    const int*   __restrict__ sf,     // (B, NV, NRINGS, NDIRS, 3)
    const float* __restrict__ ker,    // (NRINGS, NDIRS, C, NF)
    const float* __restrict__ cker,   // (C, NF)
    const float* __restrict__ bias,   // (NF,)
    float*       __restrict__ out)    // (B, NV, NF)
{
    __shared__ _Float16 Bw[NF][KTOT];                       // weights, [n][k], f16
    __shared__ _Float16 patch[WAVES][2][NRINGS][NDIRS][NC]; // wave-private gather tiles
    __shared__ _Float16 ycen[WAVES][2][NDIRS][NC];          // wave-private center rows

    const int tid  = threadIdx.x;
    const int lane = tid & 31;
    const int wave = tid >> 5;

    // ---- one-time weight prep: f32 global -> f16 LDS, [n][k], zero padded ----
    for (int idx = tid; idx < NF * KTOT; idx += 256) {
        const int n = idx / KTOT;
        const int k = idx - n * KTOT;
        float v = 0.0f;
        if (k < KCONV) {
            const int r  = k >> 7;
            const int rm = k & 127;
            const int dk = rm >> 4;
            const int c  = rm & 15;
            v = ker[((r * NDIRS + dk) * NC + c) * NF + n];
        } else if (k < KCONV + NC) {
            v = cker[(k - KCONV) * NF + n];
        }
        Bw[n][k] = (_Float16)v;
    }
    __syncthreads();

    // Fragment-lane mapping (wave32):
    //   A: M = lane&15; lanes<16 hold step-K {0..7,16..23}, lanes>=16 {8..15,24..31}
    //   B: N = lane&15; lanes<16 hold step-K {0..7,16..23}, lanes>=16 {8..15,24..31}
    const int half = lane >> 4;
    const int mrow = lane & 15;
    const int viA  = mrow >> 3;
    const int dout = mrow & 7;
    const int ncol = lane & 15;
    const int aK0  = half * 8;

    // ---- B fragments are pair-invariant: load once into registers ----
    v16h Bfrag[13];
    #pragma unroll
    for (int s = 0; s < 13; ++s) {                  // s==12 covers center+pad region
        const int bk = s * 32 + half * 8;
        const v8h blo = *(const v8h*)&Bw[ncol][bk];
        const v8h bhi = *(const v8h*)&Bw[ncol][bk + 16];
        Bfrag[s] = __builtin_shufflevector(blo, bhi,
            0, 1, 2, 3, 4, 5, 6, 7, 8, 9, 10, 11, 12, 13, 14, 15);
    }
    const float bn = bias[ncol];

    const int stride = gridDim.x * WAVES;
    for (int pair = blockIdx.x * WAVES + wave; pair < NPAIRS; pair += stride) {
        // prefetch next iteration's index block (contiguous 576 B per pair)
        const int np = pair + stride;
        if (np < NPAIRS) {
            __builtin_prefetch(sf + (long)np * (2 * NRINGS * NDIRS * 3) + lane * 4, 0, 1);
        }

        // ---- gather 2*24 random 64-B vectors: 4 lanes per vector ----
        #pragma unroll
        for (int it = 0; it < 6; ++it) {
            int task = it * 32 + lane;              // 192 = 2 * 3 * 8 * 4
            const int q  = task & 3;  task >>= 2;
            const int d  = task & 7;  task >>= 3;
            const int r  = task % 3;
            const int vi = task / 3;
            const int gv = 2 * pair + vi;           // flat (b*NV+v): no division needed
            const long base = (((long)gv * NRINGS + r) * NDIRS + d) * 3;
            const int gb = sf[base + 0];
            const int gq = sf[base + 1];
            const int gd = sf[base + 2];
            const float* src = y + (((long)gb * NV + gq) * NDIRS + gd) * NC + q * 4;
            const v4f f = *(const v4f*)src;
            const v2h plo = cvt_pk_h2(f.x, f.y);
            const v2h phi = cvt_pk_h2(f.z, f.w);
            const v4h h = __builtin_shufflevector(plo, phi, 0, 1, 2, 3);
            *(v4h*)&patch[wave][vi][r][d][q * 4] = h;
        }
        // own rows for the center term
        #pragma unroll
        for (int it = 0; it < 2; ++it) {
            int task = it * 32 + lane;              // 64 = 2 * 8 * 4
            const int q  = task & 3;  task >>= 2;
            const int d  = task & 7;
            const int vi = task >> 3;
            const int gv = 2 * pair + vi;
            const float* src = y + ((long)gv * NDIRS + d) * NC + q * 4;
            const v4f f = *(const v4f*)src;
            const v2h plo = cvt_pk_h2(f.x, f.y);
            const v2h phi = cvt_pk_h2(f.z, f.w);
            const v4h h = __builtin_shufflevector(plo, phi, 0, 1, 2, 3);
            *(v4h*)&ycen[wave][vi][d][q * 4] = h;
        }
        // patch/ycen are wave-private and LDS ops of a wave are in-order:
        // no workgroup barrier needed before reading the fragments back.

        // ---- 13 x v_wmma_f32_16x16x32_f16: M=16 (2 verts x 8 douts), N=16, K=416 ----
        v8f acc = {};
        #pragma unroll
        for (int s = 0; s < 12; ++s) {              // conv region K = 0..383
            const int gk0 = s * 32 + aK0;
            const int gk1 = gk0 + 16;
            const int r0 = gk0 >> 7, rm0 = gk0 & 127, dk0 = rm0 >> 4, c0 = rm0 & 15;
            const int r1 = gk1 >> 7, rm1 = gk1 & 127, dk1 = rm1 >> 4, c1 = rm1 & 15;
            const int d0 = (dout + dk0) & 7;        // circular padding for free
            const int d1 = (dout + dk1) & 7;
            const v8h alo = *(const v8h*)&patch[wave][viA][r0][d0][c0];
            const v8h ahi = *(const v8h*)&patch[wave][viA][r1][d1][c1];
            const v16h A = __builtin_shufflevector(alo, ahi,
                0, 1, 2, 3, 4, 5, 6, 7, 8, 9, 10, 11, 12, 13, 14, 15);
            acc = __builtin_amdgcn_wmma_f32_16x16x32_f16(
                false, A, false, Bfrag[s], (short)0, acc, false, false);
        }
        {   // center term (K 384..399) + zero pad (K 400..415)
            const v8h alo = *(const v8h*)&ycen[wave][viA][dout][aK0];
            const v8h ahi = {};
            const v16h A = __builtin_shufflevector(alo, ahi,
                0, 1, 2, 3, 4, 5, 6, 7, 8, 9, 10, 11, 12, 13, 14, 15);
            acc = __builtin_amdgcn_wmma_f32_16x16x32_f16(
                false, A, false, Bfrag[12], (short)0, acc, false, false);
        }

        // ---- epilogue: max over the 8 douts held in this lane's 8 acc VGPRs ----
        float mx = acc[0];
        #pragma unroll
        for (int j = 1; j < 8; ++j) mx = fmaxf(mx, acc[j]);
        mx = fmaxf(mx + bn, 0.0f);
        out[(2 * pair + half) * NF + ncol] = mx;    // 128-B coalesced store per wave
    }
}

extern "C" void kernel_launch(void* const* d_in, const int* in_sizes, int n_in,
                              void* d_out, int out_size, void* d_ws, size_t ws_size,
                              hipStream_t stream) {
    const float* y    = (const float*)d_in[0];
    const int*   sf   = (const int*)  d_in[1];
    const float* ker  = (const float*)d_in[2];
    const float* cker = (const float*)d_in[3];
    const float* bias = (const float*)d_in[4];
    float* out = (float*)d_out;

    hipLaunchKernelGGL(geo_conv_wmma, dim3(BLOCKS), dim3(256), 0, stream,
                       y, sf, ker, cker, bias, out);
}
